// InitialContraction_47004122087961
// MI455X (gfx1250) — matrix operand order repaired and checked by reading, not compiled, not run on hardware
//
#include <hip/hip_runtime.h>
#include <hip/hip_bf16.h>

// ---------------------------------------------------------------------------
// Fused MACE-style contraction for MI455X (gfx1250, wave32, WMMA):
//   out[b,c,p] = sum_{i,k} U[p,i,k] * (sum_e attr[b,e] W[e,k,c]) * emb[b,c,i]
// GEMM view: M=P=256, K=Y*Z=1024, N=B*C=131072; B-operand built on the fly.
// bf16 inputs / f32 accumulation via v_wmma_f32_16x16x32_bf16.
// U fragments double-buffered into LDS via global_load_async_to_lds_b128
// (ASYNCcnt), overlapping L2->LDS traffic with the WMMA stream.
// ---------------------------------------------------------------------------

typedef __attribute__((ext_vector_type(16))) __bf16 v16bf;
typedef __attribute__((ext_vector_type(8)))  float  v8f;

#define B_ATOMS   1024
#define C_CH      128
#define Y_DIM     16
#define Z_DIM     64
#define P_DIM     256          // SPARE(1,16,16) flattened
#define KTOT      1024         // Y*Z
#define KTILES    32           // KTOT / 32
#define MTILES    16           // P_DIM / 16
#define AT_STRIDE 80           // padded bf16 row stride for A^T in LDS (160B, 32B-aligned rows)

// ---------------------------------------------------------------------------
// Kernel 1: pack U (fp32, layout [p][i*64+k]) into bf16 WMMA A-fragments.
// Fragment (m,t): 32 lanes x 16 bf16, in contiguous k order (B uses the same; not ISA table order) for the
// 16x32 bf16 A tile (ISA 7.12.2):
//   row M   = lane & 15
//   K-local = j + 16*(lane>=16)
// Stored at upack[((m*32 + t)*32 + lane)*16 + j].
// ---------------------------------------------------------------------------
__global__ void u_pack_kernel(const float* __restrict__ U, __bf16* __restrict__ upack) {
    const int gid  = blockIdx.x * 256 + threadIdx.x;     // 16 * 32 * 32 = 16384 threads
    const int lane = gid & 31;
    const int t    = (gid >> 5) & 31;
    const int m    = gid >> 10;
    const int p    = m * 16 + (lane & 15);
    const int hi8  = (lane >> 4) << 4;

    const float* src = U + (size_t)p * KTOT + t * 32;
    v16bf frag;
#pragma unroll
    for (int j = 0; j < 16; ++j) {
        const int kl = j + hi8;
        frag[j] = (__bf16)src[kl];
    }
    *reinterpret_cast<v16bf*>(upack + ((size_t)((m * 32 + t) * 32 + lane)) * 16) = frag;
}

// ---------------------------------------------------------------------------
// Async memory->LDS: 64 B per lane (4 x b128). The instruction's offset
// immediate is added to BOTH the LDS (VDST) and global (VADDR) addresses
// (ISA 08 4.4), so one LDS base + one 64-bit global base covers all 4 chunks.
// Each call raises this wave's ASYNCcnt by 4; completion is in-order.
// ---------------------------------------------------------------------------
__device__ __forceinline__ void stage_async4(uint32_t lds_base, uint64_t gaddr) {
    asm volatile(
        "global_load_async_to_lds_b128 %0, %1, off\n\t"
        "global_load_async_to_lds_b128 %0, %1, off offset:16\n\t"
        "global_load_async_to_lds_b128 %0, %1, off offset:32\n\t"
        "global_load_async_to_lds_b128 %0, %1, off offset:48"
        :: "v"(lds_base), "v"(gaddr)
        : "memory");
}

// ---------------------------------------------------------------------------
// Kernel 2: one workgroup (8 waves) per atom b. Each wave owns 16 channels
// (N-tile) and the full P=256 (16 accumulator tiles). K-loop: 32 steps of 32.
// U fragments for step t+1 are async-loaded into the alternate LDS buffer
// while step t's 16 WMMAs run; B fragments are built per lane from emb and
// A^T (rank-1 outer product, never materialized).
// ---------------------------------------------------------------------------
__global__ void __launch_bounds__(256, 1)
fused_contraction_kernel(const float* __restrict__ emb,   // (B, C, Y)
                         const float* __restrict__ attr,  // (B, E)
                         const float* __restrict__ W,     // (E, Z, C)
                         const __bf16* __restrict__ upack,
                         float* __restrict__ out)         // (B, C, P)
{
    __shared__ __align__(32) __bf16 At[C_CH * AT_STRIDE];     // A^T bf16: [c][k], 20 KB
    __shared__ __align__(16) float  embS[C_CH * Y_DIM];       // 8 KB
    __shared__ __align__(64) __bf16 uStage[2][MTILES * 512];  // double-buffered K-step frags, 32 KB

    const int b    = blockIdx.x;
    const int tid  = threadIdx.x;
    const int lane = tid & 31;
    const int wave = tid >> 5;

    // ---- A[b,k,c] = sum_e attr[b,e] * W[e,k,c], stored transposed as bf16 ----
    const float a0 = attr[b * 4 + 0], a1 = attr[b * 4 + 1];
    const float a2 = attr[b * 4 + 2], a3 = attr[b * 4 + 3];
    for (int idx = tid; idx < Z_DIM * C_CH; idx += 256) {
        const int k = idx >> 7;          // 0..63
        const int c = idx & 127;         // coalesced over c
        const float v = a0 * W[0 * 8192 + k * 128 + c] + a1 * W[1 * 8192 + k * 128 + c]
                      + a2 * W[2 * 8192 + k * 128 + c] + a3 * W[3 * 8192 + k * 128 + c];
        At[c * AT_STRIDE + k] = (__bf16)v;
    }
    // ---- stage emb[b] ----
    for (int idx = tid; idx < C_CH * Y_DIM; idx += 256) {
        embS[idx] = emb[(size_t)b * (C_CH * Y_DIM) + idx];
    }

    // ---- async staging addresses (this thread owns 64 B of tile sm per step) ----
    const int sm = tid >> 4;             // m-tile staged by this thread
    const int sb = tid & 15;             // 64 B chunk within the 1 KB tile
    // global fragment bytes: (m*32 + t)*1024 ; LDS generic addr low 32 bits == LDS offset
    const uint64_t gbase = (uint64_t)(uintptr_t)upack + (size_t)sm * 32768 + (size_t)sb * 64;
    const uint32_t lds0  = (uint32_t)(uintptr_t)&uStage[0][0] + sm * 1024 + sb * 64;
    const uint32_t lds1  = (uint32_t)(uintptr_t)&uStage[1][0] + sm * 1024 + sb * 64;

    const int nl = lane & 15;            // column within this wave's c-tile
    const int hi = lane >> 4;            // half-wave selector
    const int c  = wave * 16 + nl;       // this lane's channel

    v8f acc[MTILES];
#pragma unroll
    for (int m = 0; m < MTILES; ++m) { v8f z = {}; acc[m] = z; }

    // ---- prologue: async-load K-step 0 into buffer 0 ----
    stage_async4(lds0, gbase);

    for (int t = 0; t < KTILES; ++t) {
        // issue next step's fragments into the alternate buffer (last step
        // re-issues t=31 harmlessly; drained after the loop)
        const int tn = (t + 1 < KTILES) ? (t + 1) : (KTILES - 1);
        stage_async4(((t + 1) & 1) ? lds1 : lds0, gbase + (uint64_t)tn * 1024);

        // wait for step t's 4 loads (in-order; next step's 4 stay in flight),
        // then make all waves' LDS writes visible
        asm volatile("s_wait_asynccnt 0x4" ::: "memory");
        __syncthreads();

        // ---- build B fragment: Bmat[k][n] = emb[c,i] * A[k,c] ----
        const int   i     = t >> 1;                  // kk = i*64 + k, 2 steps per i
        const int   kbase = (t & 1) * 32 + hi * 16;  // K-local 16*(lane>=16)+j
        const float e     = embS[c * Y_DIM + i];
        const v16bf arow  = *reinterpret_cast<const v16bf*>(&At[c * AT_STRIDE + kbase]);
        v16bf bfrag;
#pragma unroll
        for (int j = 0; j < 16; ++j) bfrag[j] = (__bf16)(e * (float)arow[j]);

        // ---- 16 WMMAs: full P sweep for this K-step ----
        const __bf16* ubuf = &uStage[t & 1][0];
#pragma unroll
        for (int m = 0; m < MTILES; ++m) {
            const v16bf afrag = *reinterpret_cast<const v16bf*>(ubuf + m * 512 + lane * 16);
            acc[m] = __builtin_amdgcn_wmma_f32_16x16x32_bf16(
                         /*neg_a=*/false, afrag, /*neg_b=*/false, bfrag,
                         /*c_mod=*/(short)0, acc[m], /*reuse_a=*/false, /*reuse_b=*/false);
        }

        // all waves done reading uStage[t&1] before it is refilled at t+2
        __syncthreads();
    }

    // drain the (harmless) extra prefetch before the epilogue
    asm volatile("s_wait_asynccnt 0x0" ::: "memory");

    // ---- epilogue: D tile rows p = 16m + 8*hi + r, 32B contiguous per lane ----
    float* orow = out + (size_t)b * (C_CH * P_DIM) + (size_t)c * P_DIM + hi * 8;
#pragma unroll
    for (int m = 0; m < MTILES; ++m) {
        float4* dst = reinterpret_cast<float4*>(orow + m * 16);
        float4 lo = { acc[m][0], acc[m][1], acc[m][2], acc[m][3] };
        float4 hh = { acc[m][4], acc[m][5], acc[m][6], acc[m][7] };
        dst[0] = lo;
        dst[1] = hh;
    }
}

// ---------------------------------------------------------------------------
extern "C" void kernel_launch(void* const* d_in, const int* in_sizes, int n_in,
                              void* d_out, int out_size, void* d_ws, size_t ws_size,
                              hipStream_t stream) {
    const float* emb  = (const float*)d_in[0];   // (1024, 128, 16)
    const float* attr = (const float*)d_in[1];   // (1024, 4)
    const float* U    = (const float*)d_in[2];   // (1,16,16,16,64) -> (256, 1024)
    const float* W    = (const float*)d_in[3];   // (4, 64, 128)
    float* out        = (float*)d_out;           // (1024, 128, 256)

    __bf16* upack = reinterpret_cast<__bf16*>(d_ws);   // 256*1024 bf16 = 512 KB

    // Pack U into WMMA fragment order (bf16), then run the fused GEMM.
    u_pack_kernel<<<64, 256, 0, stream>>>(U, upack);
    fused_contraction_kernel<<<B_ATOMS, 256, 0, stream>>>(emb, attr, W, upack, out);
}
